// GRU_44513041056013
// MI455X (gfx1250) — compile-verified, matrix operand order checked by reading, not run on hardware
//
#include <hip/hip_runtime.h>
#include <hip/hip_bf16.h>
#include <math.h>

// ---------------------------------------------------------------------------
// 2-layer GRU + vocab projection for MI455X (gfx1250, wave32, WMMA).
// fp32 weights are converted once per launch into bf16, pre-swizzled into the
// per-lane WMMA B-fragment layout; recurrence runs bf16 WMMA with f32 accum.
// A-operand (activations) is staged in LDS once per block and shared by all
// 8 waves, halving L2 read traffic in the dominant output projection.
// ---------------------------------------------------------------------------

typedef __attribute__((ext_vector_type(16))) __bf16 v16bf;
typedef __attribute__((ext_vector_type(8)))  __bf16 v8bf;
typedef __attribute__((ext_vector_type(8)))  float  v8f;

#define BATCH   16
#define SEQ     256
#define STATE   1024
#define EMB     1024
#define VOCAB   32000
#define LAYERS  2
#define KTOT    (EMB + STATE)          // 2048
#define NT_GATE (STATE / 16)           // 64 N-tiles for gate matmuls
#define NT_VOC  (VOCAB / 16)           // 2000 N-tiles for output projection
#define KT_VOC  (STATE / 32)           // 32 K-tiles for output projection

// Packed B-fragment matrix sizes (elements)
static constexpr size_t GATE_MAT_ELEMS = (size_t)NT_GATE * (KTOT / 32) * 32 * 16; // 2,097,152 = 1<<21
static constexpr size_t WO_ELEMS       = (size_t)NT_VOC * KT_VOC * 32 * 16;       // 32,768,000
static constexpr size_t EMB_ELEMS      = (size_t)SEQ * BATCH * EMB;               // 4,194,304

// Workspace layout (byte offsets)
static constexpr size_t OFF_GATES = 0;
static constexpr size_t OFF_WO    = OFF_GATES + 6 * GATE_MAT_ELEMS * 2;   // 25,165,824
static constexpr size_t OFF_EMB   = OFF_WO + WO_ELEMS * 2;                // + 65,536,000
static constexpr size_t OFF_H     = OFF_EMB + EMB_ELEMS * 2;              // + 8,388,608
static constexpr size_t OFF_HBF   = OFF_H  + (size_t)LAYERS * BATCH * STATE * 4;
static constexpr size_t OFF_Z     = OFF_HBF + (size_t)LAYERS * BATCH * STATE * 2;
static constexpr size_t OFF_RH    = OFF_Z  + (size_t)BATCH * STATE * 4;

// ---------------------------------------------------------------------------
// Fragment loaders.
// A (16x32 bf16, row-major src):
//   lane<16 : row=lane,    K = {0..7, 16..23}
//   lane>=16: row=lane-16, K = {8..15, 24..31}
// B packed layout mirrors this with lane->column; conversion kernels below
// write the weights so each lane reads 16 contiguous bf16 (2x b128).
// ---------------------------------------------------------------------------
__device__ __forceinline__ v16bf cat8(v8bf lo, v8bf hi) {
    return __builtin_shufflevector(lo, hi, 0,1,2,3,4,5,6,7,8,9,10,11,12,13,14,15);
}
__device__ __forceinline__ v16bf load_a_frag(const __bf16* p) {   // LDS after inlining
    v8bf lo = *(const v8bf*)p;
    v8bf hi = *(const v8bf*)(p + 16);
    return cat8(lo, hi);
}
__device__ __forceinline__ v16bf load_b_frag(const __bf16* p) {   // global packed
    v8bf lo = *(const v8bf*)p;
    v8bf hi = *(const v8bf*)(p + 8);
    return cat8(lo, hi);
}
__device__ __forceinline__ v8f wmma_bf16(v16bf a, v16bf b, v8f c) {
    return __builtin_amdgcn_wmma_f32_16x16x32_bf16(false, a, false, b, (short)0, c, false, false);
}
__device__ __forceinline__ float sigmoidf_fast(float x) {
    return 1.0f / (1.0f + __expf(-x));
}

// ---------------------------------------------------------------------------
// One-time per-launch preparation kernels
// ---------------------------------------------------------------------------
__global__ void init_state_kernel(float* __restrict__ h, __bf16* __restrict__ hbf) {
    int idx = blockIdx.x * blockDim.x + threadIdx.x;  // LAYERS*BATCH*STATE = 32768
    h[idx]   = 0.0f;
    hbf[idx] = (__bf16)0.0f;
}

// Pack 6 gate weight matrices [K=2048, N=1024] fp32 -> bf16 B-fragments.
// dst index = [mat(6)][ntile(64)][ktile(64)][lane(32)][e(16)]
__global__ void pack_gates_kernel(const float* __restrict__ Wz,
                                  const float* __restrict__ Wr,
                                  const float* __restrict__ Wn,
                                  __bf16* __restrict__ dst) {
    size_t idx = (size_t)blockIdx.x * blockDim.x + threadIdx.x;  // 12,582,912 total
    int e    = (int)(idx & 15);
    int lane = (int)((idx >> 4) & 31);
    int kt   = (int)((idx >> 9) & 63);
    int nt   = (int)((idx >> 15) & 63);
    int mat  = (int)(idx >> 21);           // 0..5
    int layer = mat / 3, gate = mat % 3;
    const float* W = (gate == 0) ? Wz : (gate == 1) ? Wr : Wn;
    W += (size_t)layer * KTOT * STATE;
    int klocal = (e & 7) + ((lane >> 4) << 3) + ((e >> 3) << 4);
    int n = nt * 16 + (lane & 15);
    int k = kt * 32 + klocal;
    dst[idx] = (__bf16)W[(size_t)k * STATE + n];
}

// Pack Wo [K=1024, N=32000] fp32 -> bf16 B-fragments.
// dst index = [ntile(2000)][ktile(32)][lane(32)][e(16)]
__global__ void pack_wo_kernel(const float* __restrict__ Wo, __bf16* __restrict__ dst) {
    size_t idx = (size_t)blockIdx.x * blockDim.x + threadIdx.x;  // 32,768,000 total
    int e    = (int)(idx & 15);
    int lane = (int)((idx >> 4) & 31);
    int kt   = (int)((idx >> 9) & 31);
    int nt   = (int)(idx >> 14);           // 0..1999
    int klocal = (e & 7) + ((lane >> 4) << 3) + ((e >> 3) << 4);
    int n = nt * 16 + (lane & 15);
    int k = kt * 32 + klocal;
    dst[idx] = (__bf16)Wo[(size_t)k * VOCAB + n];
}

// Gather + convert embeddings: dst[t][b][c] = bf16(emb[x[b,t]][c])
__global__ void embed_kernel(const int* __restrict__ x,
                             const float* __restrict__ emb,
                             __bf16* __restrict__ dst) {
    size_t idx = (size_t)blockIdx.x * blockDim.x + threadIdx.x;  // 4,194,304 total
    int c = (int)(idx & 1023);
    int b = (int)((idx >> 10) & 15);
    int t = (int)(idx >> 14);
    int row = x[b * SEQ + t];
    dst[idx] = (__bf16)emb[(size_t)row * EMB + c];
}

// ---------------------------------------------------------------------------
// Per-timestep GRU kernels. One wave handles one 16-column N-tile.
// Block = 256 threads = 8 waves = 8 N-tiles; A staged once per block in LDS.
// ---------------------------------------------------------------------------

// Cooperatively fill sA[16][KTOT] from two row-major [16][1024] halves.
__device__ __forceinline__ void stage_a2(__bf16* sA, const __bf16* h0, const __bf16* h1) {
    v8bf* dst = (v8bf*)sA;
    for (int ci = threadIdx.x; ci < (BATCH * KTOT) / 8; ci += 256) {
        int r = ci >> 8;                 // 256 v8bf chunks per 2048-wide row
        int c = (ci & 255) * 8;
        const __bf16* src = (c < EMB) ? (h0 + r * EMB + c)
                                      : (h1 + r * STATE + (c - EMB));
        dst[ci] = *(const v8bf*)src;
    }
    __syncthreads();
}

// z = sigmoid([inp,h] @ Wz + bz); r = sigmoid([inp,h] @ Wr + br)
// writes z (f32) and r*h (bf16) for the n-gate matmul.
__global__ __launch_bounds__(256) void gru_zr_kernel(
    const __bf16* __restrict__ inp,      // [16,1024] bf16 (layer input)
    const __bf16* __restrict__ hbf,      // [16,1024] bf16 (recurrent state)
    const float*  __restrict__ hstate,   // [16,1024] f32
    const __bf16* __restrict__ packZ,
    const __bf16* __restrict__ packR,
    const float*  __restrict__ bz,
    const float*  __restrict__ br,
    float*        __restrict__ zbuf,     // [16,1024] f32
    __bf16*       __restrict__ rhbuf) {  // [16,1024] bf16
    __shared__ __align__(16) __bf16 sA[BATCH * KTOT];   // 64 KB
    stage_a2(sA, inp, hbf);

    int lane = threadIdx.x & 31;
    int nt   = blockIdx.x * 8 + (threadIdx.x >> 5);   // 0..63
    int row  = lane & 15;
    int sub  = (lane >> 4) << 3;

    const __bf16* pz   = packZ + (size_t)nt * 64 * 512 + lane * 16;
    const __bf16* pr   = packR + (size_t)nt * 64 * 512 + lane * 16;
    const __bf16* aRow = sA + row * KTOT + sub;

    v8f cz = {}; v8f cr = {};
    for (int kt = 0; kt < 64; ++kt) {
        v16bf a  = load_a_frag(aRow + kt * 32);
        v16bf wz = load_b_frag(pz + kt * 512);
        v16bf wr = load_b_frag(pr + kt * 512);
        cz = wmma_bf16(a, wz, cz);
        cr = wmma_bf16(a, wr, cr);
    }

    int col = nt * 16 + (lane & 15);
    int mbase = (lane >> 4) << 3;
    float bzv = bz[col], brv = br[col];
#pragma unroll
    for (int v = 0; v < 8; ++v) {
        int i = (mbase + v) * STATE + col;
        float z = sigmoidf_fast(cz[v] + bzv);
        float r = sigmoidf_fast(cr[v] + brv);
        zbuf[i]  = z;
        rhbuf[i] = (__bf16)(r * hstate[i]);
    }
}

// n = tanh([inp, r*h] @ Wn + bn); h' = (1-z)*h + z*n; writes f32 + bf16 state.
__global__ __launch_bounds__(256) void gru_nh_kernel(
    const __bf16* __restrict__ inp,
    const __bf16* __restrict__ rhbuf,
    float*        __restrict__ hstate,   // read+write in place (per-element)
    __bf16*       __restrict__ hbf,      // bf16 copy of new state
    const __bf16* __restrict__ packN,
    const float*  __restrict__ bn,
    const float*  __restrict__ zbuf) {
    __shared__ __align__(16) __bf16 sA[BATCH * KTOT];   // 64 KB
    stage_a2(sA, inp, rhbuf);

    int lane = threadIdx.x & 31;
    int nt   = blockIdx.x * 8 + (threadIdx.x >> 5);
    int row  = lane & 15;
    int sub  = (lane >> 4) << 3;

    const __bf16* pn   = packN + (size_t)nt * 64 * 512 + lane * 16;
    const __bf16* aRow = sA + row * KTOT + sub;

    v8f cn = {};
    for (int kt = 0; kt < 64; ++kt) {
        v16bf a  = load_a_frag(aRow + kt * 32);
        v16bf wn = load_b_frag(pn + kt * 512);
        cn = wmma_bf16(a, wn, cn);
    }

    int col = nt * 16 + (lane & 15);
    int mbase = (lane >> 4) << 3;
    float bnv = bn[col];
#pragma unroll
    for (int v = 0; v < 8; ++v) {
        int i = (mbase + v) * STATE + col;
        float n  = tanhf(cn[v] + bnv);
        float z  = zbuf[i];
        float hn = (1.0f - z) * hstate[i] + z * n;
        hstate[i] = hn;
        hbf[i]    = (__bf16)hn;
    }
}

// out[b, t, :] = h_layer1 @ Wo + bo
__global__ __launch_bounds__(256) void out_proj_kernel(
    const __bf16* __restrict__ hbf,      // [16,1024] bf16, last layer
    const __bf16* __restrict__ packWo,
    const float*  __restrict__ bo,
    float*        __restrict__ out,      // [16, 256, 32000] f32
    int t) {
    __shared__ __align__(16) __bf16 sA[BATCH * STATE];  // 32 KB
    {
        v8bf* dst = (v8bf*)sA;
        const v8bf* src = (const v8bf*)hbf;
        for (int ci = threadIdx.x; ci < (BATCH * STATE) / 8; ci += 256)
            dst[ci] = src[ci];
        __syncthreads();
    }

    int lane = threadIdx.x & 31;
    int nt   = blockIdx.x * 8 + (threadIdx.x >> 5);   // 0..1999
    int row  = lane & 15;
    int sub  = (lane >> 4) << 3;

    const __bf16* pw   = packWo + (size_t)nt * KT_VOC * 512 + lane * 16;
    const __bf16* aRow = sA + row * STATE + sub;

    v8f c = {};
    for (int kt = 0; kt < KT_VOC; ++kt) {
        v16bf a = load_a_frag(aRow + kt * 32);
        v16bf w = load_b_frag(pw + kt * 512);
        c = wmma_bf16(a, w, c);
    }

    int col = nt * 16 + (lane & 15);
    int mbase = (lane >> 4) << 3;
    float bov = bo[col];
#pragma unroll
    for (int v = 0; v < 8; ++v) {
        int b = mbase + v;
        out[((size_t)b * SEQ + t) * VOCAB + col] = c[v] + bov;
    }
}

// Copy final hidden state [LAYERS, BATCH, STATE] after the outputs.
__global__ void copy_final_kernel(const float* __restrict__ hstate, float* __restrict__ out) {
    int idx = blockIdx.x * blockDim.x + threadIdx.x;  // 32768
    out[(size_t)BATCH * SEQ * VOCAB + idx] = hstate[idx];
}

// ---------------------------------------------------------------------------
// Host launcher
// ---------------------------------------------------------------------------
extern "C" void kernel_launch(void* const* d_in, const int* in_sizes, int n_in,
                              void* d_out, int out_size, void* d_ws, size_t ws_size,
                              hipStream_t stream) {
    (void)in_sizes; (void)n_in; (void)out_size; (void)ws_size;

    const int*   x   = (const int*)  d_in[0];
    const float* emb = (const float*)d_in[1];
    const float* Wz  = (const float*)d_in[2];
    const float* bz  = (const float*)d_in[3];
    const float* Wr  = (const float*)d_in[4];
    const float* br  = (const float*)d_in[5];
    const float* Wn  = (const float*)d_in[6];
    const float* bn  = (const float*)d_in[7];
    const float* Wo  = (const float*)d_in[8];
    const float* bo  = (const float*)d_in[9];
    float* outF = (float*)d_out;

    char* ws = (char*)d_ws;
    __bf16* gates  = (__bf16*)(ws + OFF_GATES);
    __bf16* woPk   = (__bf16*)(ws + OFF_WO);
    __bf16* embBf  = (__bf16*)(ws + OFF_EMB);
    float*  hstate = (float*) (ws + OFF_H);
    __bf16* hbf    = (__bf16*)(ws + OFF_HBF);
    float*  zbuf   = (float*) (ws + OFF_Z);
    __bf16* rhbuf  = (__bf16*)(ws + OFF_RH);

    // One-time prep (inside the graph: deterministic, runs every call)
    init_state_kernel<<<(LAYERS * BATCH * STATE) / 256, 256, 0, stream>>>(hstate, hbf);
    pack_gates_kernel<<<(unsigned)((6 * GATE_MAT_ELEMS) / 256), 256, 0, stream>>>(Wz, Wr, Wn, gates);
    pack_wo_kernel<<<(unsigned)(WO_ELEMS / 256), 256, 0, stream>>>(Wo, woPk);
    embed_kernel<<<(unsigned)(EMB_ELEMS / 256), 256, 0, stream>>>(x, emb, embBf);

    // Recurrence: stream ordering provides all grid-wide dependencies.
    for (int t = 0; t < SEQ; ++t) {
        for (int j = 0; j < LAYERS; ++j) {
            const __bf16* inp = (j == 0) ? (embBf + (size_t)t * BATCH * EMB)
                                         : (hbf + (size_t)(j - 1) * BATCH * STATE);
            float*  hj   = hstate + (size_t)j * BATCH * STATE;
            __bf16* hbfj = hbf    + (size_t)j * BATCH * STATE;
            const __bf16* pZ = gates + (size_t)(j * 3 + 0) * GATE_MAT_ELEMS;
            const __bf16* pR = gates + (size_t)(j * 3 + 1) * GATE_MAT_ELEMS;
            const __bf16* pN = gates + (size_t)(j * 3 + 2) * GATE_MAT_ELEMS;

            gru_zr_kernel<<<NT_GATE / 8, 256, 0, stream>>>(
                inp, hbfj, hj, pZ, pR, bz + j * STATE, br + j * STATE, zbuf, rhbuf);
            gru_nh_kernel<<<NT_GATE / 8, 256, 0, stream>>>(
                inp, rhbuf, hj, hbfj, pN, bn + j * STATE, zbuf);
        }
        out_proj_kernel<<<NT_VOC / 8, 256, 0, stream>>>(
            hbf + (size_t)(LAYERS - 1) * BATCH * STATE, woPk, bo, outF, t);
    }

    copy_final_kernel<<<(LAYERS * BATCH * STATE) / 256, 256, 0, stream>>>(hstate, outF);
}
